// LSTM_10960756539796
// MI455X (gfx1250) — 8-bit matrix operand order repaired and checked by reading, compiled, not run on hardware
//
#include <hip/hip_runtime.h>
#include <math.h>


typedef int   v8i  __attribute__((ext_vector_type(8)));
typedef int   v16i __attribute__((ext_vector_type(16)));
typedef float v8f  __attribute__((ext_vector_type(8)));

#define TT 512
#define DD 64
#define HH 256
#define G4 1024
#define ROWS 16
#define NWG  16
#define HSTRIDE 264
#define WLDS_BYTES (G4 * HH)
#define H8_BYTES   (ROWS * HSTRIDE)
#define X8_OFF     WLDS_BYTES

__device__ __forceinline__ unsigned char f2fp8(float x) {
  return (unsigned char)(__builtin_amdgcn_cvt_pk_fp8_f32(x, x, 0, false) & 0xff);
}
__device__ __forceinline__ int pk4_fp8(float a, float b, float c, float d) {
  int lo = __builtin_amdgcn_cvt_pk_fp8_f32(a, b, 0, false);
  return __builtin_amdgcn_cvt_pk_fp8_f32(c, d, lo, true);
}
__device__ __forceinline__ float ftanh(float x) {
#if __has_builtin(__builtin_amdgcn_tanhf)
  return __builtin_amdgcn_tanhf(x);
#else
  float e = __expf(2.0f * x);
  return 1.0f - 2.0f / (e + 1.0f);
#endif
}
__device__ __forceinline__ float sigf(float x) {
  return 0.5f * ftanh(0.5f * x) + 0.5f;
}

__global__ void lstm_prep_whh(const float* __restrict__ W_hh,
                              unsigned char* __restrict__ ws8) {
  int n = blockIdx.x;
  int k = threadIdx.x;
  float wv = W_hh[n * HH + k] * 16.0f;
  int kr = k & 63;
  int hi = (kr >> 4) & 1;
  int v  = ((kr >= 32) ? 4 : 0) + ((kr & 15) >> 2);
  int b  = kr & 3;
  int lane  = (n & 15) + 16 * hi;
  int block = ((n >> 4) << 2) + (k >> 6);
  ws8[block * 1024 + lane * 32 + v * 4 + b] = f2fp8(wv);
}

__global__ void lstm_prep_x8(const float* __restrict__ x,
                             unsigned char* __restrict__ ws8) {
  int t  = blockIdx.x;
  int bt = blockIdx.y;
  int tid = threadIdx.x;
  int l = tid & 31, v = tid >> 5;
  int hi = l >> 4;
  int row = bt * ROWS + (l & 15);
  int koff = ((v >> 1) * 16) + hi * 8 + (v & 1) * 4;
  const float4 xv = *(const float4*)(x + ((size_t)row * TT + t) * DD + koff);
  size_t base = X8_OFF + ((size_t)(bt * TT + t)) * 1024;
  ((int*)(ws8 + base))[l * 8 + v] =
      pk4_fp8(xv.x * 8.f, xv.y * 8.f, xv.z * 8.f, xv.w * 8.f);
}

__global__ void __launch_bounds__(256, 1)
lstm_persistent(const float* __restrict__ W_ih,
                const float* __restrict__ b_ih,
                const float* __restrict__ b_hh,
                const float* __restrict__ W_out,
                const float* __restrict__ b_out,
                const unsigned char* __restrict__ ws8,
                float* __restrict__ out) {
  extern __shared__ char smem[];
  char* Wl  = smem;
  char* hb  = smem + WLDS_BYTES;

  const int tid = threadIdx.x;
  const int w   = tid >> 5;
  const int l   = tid & 31;
  const int col = l & 15;
  const int hi  = l >> 4;
  const int b0  = blockIdx.x * ROWS;
  const int row = col;

  {
    const int4* src = (const int4*)ws8;
    int4* dst = (int4*)Wl;
    for (int i = tid; i < WLDS_BYTES / 16; i += 256) dst[i] = src[i];
  }
  for (int i = tid; i < 2 * H8_BYTES / 4; i += 256) ((int*)hb)[i] = 0;

  v8i   wih[8];
  float bias[8];
#pragma unroll
  for (int t8 = 0; t8 < 8; ++t8) {
    int g = t8 >> 1, j = t8 & 1;
    int n = g * 256 + w * 32 + j * 16 + col;
    bias[t8] = b_ih[n] + b_hh[n];
#pragma unroll
    for (int v = 0; v < 8; ++v) {
      int K = ((v >> 1) * 16) + hi * 8 + (v & 1) * 4;
      const float4 wv = *(const float4*)(W_ih + n * DD + K);
      wih[t8][v] = pk4_fp8(wv.x * 16.f, wv.y * 16.f, wv.z * 16.f, wv.w * 16.f);
    }
  }

  float cst[2][8];
  float hfin[2][8];
#pragma unroll
  for (int j = 0; j < 2; ++j)
#pragma unroll
    for (int r = 0; r < 8; ++r) { cst[j][r] = 0.f; hfin[j][r] = 0.f; }

  __syncthreads();

  const unsigned char* x8 =
      ws8 + X8_OFF + (size_t)blockIdx.x * TT * 1024 + l * 32;

  for (int t = 0; t < TT; ++t) {
    const char* rb = hb + (t & 1) * H8_BYTES;
    char*       wb = hb + ((t + 1) & 1) * H8_BYTES;

    const int4* xp = (const int4*)(x8 + t * 1024);
    int4 xlo = xp[0], xhi = xp[1];
    v8i xa;
    xa[0] = xlo.x; xa[1] = xlo.y; xa[2] = xlo.z; xa[3] = xlo.w;
    xa[4] = xhi.x; xa[5] = xhi.y; xa[6] = xhi.z; xa[7] = xhi.w;
    int tp = (t + 8 < TT) ? (t + 8) : (TT - 1);
    __builtin_prefetch(x8 + tp * 1024, 0, 1);

    v8f acc[8];
#pragma unroll
    for (int t8 = 0; t8 < 8; ++t8) {
      v8f z = {};
      acc[t8] = __builtin_amdgcn_wmma_f32_16x16x64_fp8_fp8(
          xa, wih[t8], (short)0, z, false, false);
    }

#pragma unroll
    for (int kb2 = 0; kb2 < 2; ++kb2) {
      v16i ha;
#pragma unroll
      for (int half = 0; half < 2; ++half) {
        int kbase = kb2 * 128 + half * 64;
        const char* hp = rb + row * HSTRIDE + kbase + hi * 16;
#pragma unroll
        for (int p = 0; p < 4; ++p) {
          int2 d = *(const int2*)(hp + (p >> 1) * 32 + (p & 1) * 8);
          ha[half * 8 + p * 2 + 0] = d.x;
          ha[half * 8 + p * 2 + 1] = d.y;
        }
      }
#pragma unroll
      for (int t8 = 0; t8 < 8; ++t8) {
        int g = t8 >> 1, j = t8 & 1;
        int n0 = g * 16 + w * 2 + j;
        const char* bp = Wl + (n0 * 4 + kb2 * 2) * 1024 + l * 32;
        v16i bb;
#pragma unroll
        for (int q = 0; q < 4; ++q) {
          int4 d = *(const int4*)(bp + (q >> 1) * 1024 + (q & 1) * 16);
          bb[q * 4 + 0] = d.x; bb[q * 4 + 1] = d.y;
          bb[q * 4 + 2] = d.z; bb[q * 4 + 3] = d.w;
        }
        acc[t8] = __builtin_amdgcn_wmma_f32_16x16x128_fp8_fp8(
            ha, bb, (short)0, acc[t8], false, false);
      }
    }

#pragma unroll
    for (int j = 0; j < 2; ++j) {
#pragma unroll
      for (int r = 0; r < 8; ++r) {
        float zi = acc[0 * 2 + j][r] * 0.0078125f + bias[0 * 2 + j];
        float zf = acc[1 * 2 + j][r] * 0.0078125f + bias[1 * 2 + j];
        float zg = acc[2 * 2 + j][r] * 0.0078125f + bias[2 * 2 + j];
        float zo = acc[3 * 2 + j][r] * 0.0078125f + bias[3 * 2 + j];
        float cn = sigf(zf) * cst[j][r] + sigf(zi) * ftanh(zg);
        cst[j][r] = cn;
        float hn = sigf(zo) * ftanh(cn);
        hfin[j][r] = hn;
        int M    = r + (hi ? 8 : 0);
        int hcol = w * 32 + j * 16 + col;
        wb[M * HSTRIDE + hcol] = f2fp8(hn * 8.f);
      }
    }
    __syncthreads();
  }

  float* hf = (float*)smem;
#pragma unroll
  for (int j = 0; j < 2; ++j)
#pragma unroll
    for (int r = 0; r < 8; ++r) {
      int M = r + (hi ? 8 : 0);
      hf[M * HH + (w * 32 + j * 16 + col)] = hfin[j][r];
    }
  __syncthreads();

  for (int oi = tid; oi < ROWS * 10; oi += 256) {
    int rr = oi / 10, cls = oi % 10;
    float s = b_out[cls];
    const float* hr = hf + rr * HH;
    const float* wr = W_out + cls * HH;
    for (int k2 = 0; k2 < HH; ++k2) s += hr[k2] * wr[k2];
    out[(b0 + rr) * 10 + cls] = sigf(s);
  }
}

extern "C" void kernel_launch(void* const* d_in, const int* in_sizes, int n_in,
                              void* d_out, int out_size, void* d_ws, size_t ws_size,
                              hipStream_t stream) {
  const float* x     = (const float*)d_in[0];
  const float* W_ih  = (const float*)d_in[1];
  const float* W_hh  = (const float*)d_in[2];
  const float* b_ih  = (const float*)d_in[3];
  const float* b_hh  = (const float*)d_in[4];
  const float* W_out = (const float*)d_in[5];
  const float* b_out = (const float*)d_in[6];
  unsigned char* ws8 = (unsigned char*)d_ws;

  lstm_prep_whh<<<G4, HH, 0, stream>>>(W_hh, ws8);
  lstm_prep_x8<<<dim3(TT, NWG), 256, 0, stream>>>(x, ws8);

  size_t smem = WLDS_BYTES + 2 * H8_BYTES;
  lstm_persistent<<<NWG, 256, smem, stream>>>(W_ih, b_ih, b_hh,
                                              W_out, b_out, ws8, (float*)d_out);
}
